// MultiQueryAttention_13649406067535
// MI455X (gfx1250) — matrix operand order repaired and checked by reading, not compiled, not run on hardware
//
#include <hip/hip_runtime.h>
#include <stdint.h>

// ---------------------------------------------------------------------------
// MQA block for MI455X (gfx1250, wave32, WMMA bf16 + TDM data movement)
// E=2048, H=16, D=128, B=4, N=2048  -> M = B*N = 8192 rows
// ---------------------------------------------------------------------------

typedef __attribute__((ext_vector_type(16))) __bf16 v16bf;
typedef __attribute__((ext_vector_type(8)))  float  v8f;
typedef __attribute__((ext_vector_type(4)))  unsigned int v4u;
typedef __attribute__((ext_vector_type(8)))  int v8i;
typedef __attribute__((ext_vector_type(4)))  int v4i;

struct __align__(16) U4 { unsigned int x, y, z, w; };
union Frag16 { v16bf v; U4 u[2]; };

__device__ inline unsigned short f2bf(float f) {
    union { float f; unsigned int u; } c; c.f = f;
    unsigned int r = c.u + 0x7FFFu + ((c.u >> 16) & 1u);  // round-nearest-even
    return (unsigned short)(r >> 16);
}

__device__ inline v16bf load_frag(const unsigned short* p0, const unsigned short* p1) {
    Frag16 f;
    f.u[0] = *(const U4*)p0;
    f.u[1] = *(const U4*)p1;
    return f.v;
}

__device__ inline v8f wmma_bf16(v16bf a, v16bf b, v8f c) {
    return __builtin_amdgcn_wmma_f32_16x16x32_bf16(
        false, a, false, b, (short)0, c, false, false);
}

// ---------------------------------------------------------------------------
// TDM: DMA a (128 rows x 32 cols) bf16 tile from a row-major [d1][d0] tensor
// into LDS, padding each 64B row with 16B -> 80B pitch (= 40 bf16 elements),
// exactly matching the WMMA fragment addressing below.
// ---------------------------------------------------------------------------
__device__ inline void tdm_load_tile(unsigned lds_addr, const unsigned short* gptr,
                                     unsigned tdim0, unsigned tdim1, unsigned stride)
{
    unsigned long long ga = (unsigned long long)(uintptr_t)gptr;
    v4u g0;
    g0[0] = 1u;                                        // count=1, user descriptor
    g0[1] = lds_addr;                                  // LDS byte address
    g0[2] = (unsigned)(ga & 0xFFFFFFFFu);              // global addr lo
    g0[3] = (unsigned)((ga >> 32) & 0x01FFFFFFu)       // global addr hi (57-bit)
          | (2u << 30);                                // type = 2 ("image")
    v8i g1;
    g1[0] = (int)((1u << 16)                           // data_size = 2 bytes
                | (1u << 20)                           // pad_enable
                | (3u << 22)                           // pad_interval: 16 DWORDs (64B)
                | (3u << 25));                         // pad_amount: 4 DWORDs (16B)
    g1[1] = (int)((tdim0 & 0xFFFFu) << 16);            // tensor_dim0 [15:0]
    g1[2] = (int)(((tdim0 >> 16) & 0xFFFFu)            // tensor_dim0 [31:16]
                | ((tdim1 & 0xFFFFu) << 16));          // tensor_dim1 [15:0]
    g1[3] = (int)(((tdim1 >> 16) & 0xFFFFu)            // tensor_dim1 [31:16]
                | (32u << 16));                        // tile_dim0 = 32 elements
    g1[4] = 128;                                       // tile_dim1 = 128 rows
    g1[5] = (int)stride;                               // tensor_dim0_stride (elems)
    g1[6] = 0;
    g1[7] = 0;
    v4i z4 = {0, 0, 0, 0};
#if defined(__clang_major__) && (__clang_major__ >= 23)
    v8i z8 = {0, 0, 0, 0, 0, 0, 0, 0};
    __builtin_amdgcn_tensor_load_to_lds(g0, g1, z4, z4, z8, 0);
#else
    __builtin_amdgcn_tensor_load_to_lds(g0, g1, z4, z4, 0);
#endif
}

// ---------------------------------------------------------------------------
// Prepass: f32 -> bf16 elementwise (4 per thread)
// ---------------------------------------------------------------------------
__global__ __launch_bounds__(256)
void cvt_bf16(const float* __restrict__ src, unsigned short* __restrict__ dst)
{
    size_t i = ((size_t)blockIdx.x * 256 + threadIdx.x) * 4;
    float4 v = *(const float4*)(src + i);
    dst[i + 0] = f2bf(v.x);
    dst[i + 1] = f2bf(v.y);
    dst[i + 2] = f2bf(v.z);
    dst[i + 3] = f2bf(v.w);
}

// ---------------------------------------------------------------------------
// Prepass: transpose f32 [rows][cols] -> bf16 [cols][rows]
// ---------------------------------------------------------------------------
__global__ __launch_bounds__(256)
void transpose_bf16(const float* __restrict__ src, unsigned short* __restrict__ dst,
                    int rows, int cols)
{
    __shared__ float t[32][33];
    int c0 = blockIdx.x * 32;
    int r0 = blockIdx.y * 32;
    int tx = threadIdx.x & 31;
    int ty = threadIdx.x >> 5;   // 0..7
    #pragma unroll
    for (int i = 0; i < 32; i += 8)
        t[ty + i][tx] = src[(size_t)(r0 + ty + i) * cols + c0 + tx];
    __syncthreads();
    #pragma unroll
    for (int i = 0; i < 32; i += 8)
        dst[(size_t)(c0 + ty + i) * rows + r0 + tx] = f2bf(t[tx][ty + i]);
}

// ---------------------------------------------------------------------------
// GEMM:  C[M x Ncols] = A[M x K] * Bt[Ncols x K]^T + bias   (all bf16 inputs)
// WG tile 128x128, 256 threads = 8 waves (2x4); each wave 64x32 (4x2 WMMA).
// A/B tiles DMA'd by the Tensor Data Mover, double-buffered:
//   wave0 issues A-tile TDM, wave1 issues B-tile TDM for kt+1, then waits
//   s_wait_tensorcnt 1 for kt's tile while the next DMA runs under the WMMAs.
// SPLIT: scatter QKV outputs as bf16 q/k/v, else f32 + bias to outF.
// ---------------------------------------------------------------------------
template<bool SPLIT>
__global__ __launch_bounds__(256)
void gemm_wmma_tdm(const unsigned short* __restrict__ A,
                   const unsigned short* __restrict__ Bt,
                   const float* __restrict__ bias, int K, int Ncols,
                   float* __restrict__ outF,
                   unsigned short* __restrict__ qb, unsigned short* __restrict__ kb,
                   unsigned short* __restrict__ vb)
{
    __shared__ __align__(16) unsigned short As[2][128 * 40];
    __shared__ __align__(16) unsigned short Bs[2][128 * 40];

    const int tid  = threadIdx.x;
    const int lane = tid & 31;
    const int wv   = tid >> 5;   // 0..7
    const int wm   = wv >> 2;    // 0..1
    const int wn   = wv & 3;     // 0..3
    const int half = lane >> 4;
    const int l16  = lane & 15;

    const int n0 = blockIdx.x * 128;
    const int m0 = blockIdx.y * 128;

    const unsigned short* Ag = A  + (size_t)m0 * K;   // tile base, advance by 32/kt
    const unsigned short* Bg = Bt + (size_t)n0 * K;

    v8f acc[4][2];
    #pragma unroll
    for (int i = 0; i < 4; ++i)
        #pragma unroll
        for (int j = 0; j < 2; ++j)
            #pragma unroll
            for (int e = 0; e < 8; ++e) acc[i][j][e] = 0.0f;

    const int nK = K >> 5;

    // prologue: DMA tile 0 into buffer 0
    if (wv == 0)
        tdm_load_tile((unsigned)(uintptr_t)&As[0][0], Ag, (unsigned)K, 8192u, (unsigned)K);
    else if (wv == 1)
        tdm_load_tile((unsigned)(uintptr_t)&Bs[0][0], Bg, (unsigned)K, (unsigned)Ncols, (unsigned)K);

    for (int kt = 0; kt < nK; ++kt) {
        const int cur = kt & 1;
        if (wv == 0) {
            if (kt + 1 < nK) {
                tdm_load_tile((unsigned)(uintptr_t)&As[cur ^ 1][0],
                              Ag + (kt + 1) * 32, (unsigned)K, 8192u, (unsigned)K);
                __builtin_amdgcn_s_wait_tensorcnt((short)1);
            } else {
                __builtin_amdgcn_s_wait_tensorcnt((short)0);
            }
        } else if (wv == 1) {
            if (kt + 1 < nK) {
                tdm_load_tile((unsigned)(uintptr_t)&Bs[cur ^ 1][0],
                              Bg + (kt + 1) * 32, (unsigned)K, (unsigned)Ncols, (unsigned)K);
                __builtin_amdgcn_s_wait_tensorcnt((short)1);
            } else {
                __builtin_amdgcn_s_wait_tensorcnt((short)0);
            }
        }
        __syncthreads();   // tile `cur` ready for all waves

        v16bf bfr[2];
        #pragma unroll
        for (int ns = 0; ns < 2; ++ns) {
            const unsigned short* p = &Bs[cur][(wn * 32 + ns * 16 + l16) * 40 + half * 8];
            bfr[ns] = load_frag(p, p + 16);
        }
        #pragma unroll
        for (int ms = 0; ms < 4; ++ms) {
            const unsigned short* p = &As[cur][(wm * 64 + ms * 16 + l16) * 40 + half * 8];
            v16bf af = load_frag(p, p + 16);
            acc[ms][0] = wmma_bf16(af, bfr[0], acc[ms][0]);
            acc[ms][1] = wmma_bf16(af, bfr[1], acc[ms][1]);
        }
        __syncthreads();   // all reads of buffer `cur` done before it is re-DMA'd
    }

    // ---- epilogue: bias + store ----
    #pragma unroll
    for (int ns = 0; ns < 2; ++ns) {
        int ncol = n0 + wn * 32 + ns * 16 + l16;
        float bv = bias[ncol];
        #pragma unroll
        for (int ms = 0; ms < 4; ++ms) {
            int mbase = m0 + wm * 64 + ms * 16 + half * 8;
            #pragma unroll
            for (int r = 0; r < 8; ++r) {
                float val = acc[ms][ns][r] + bv;
                int mrow = mbase + r;
                if (SPLIT) {
                    if (ncol < 2048)
                        qb[(size_t)mrow * 2048 + ncol] = f2bf(val);
                    else if (ncol < 2176)
                        kb[(size_t)mrow * 128 + (ncol - 2048)] = f2bf(val);
                    else
                        vb[(size_t)mrow * 128 + (ncol - 2176)] = f2bf(val);
                } else {
                    outF[(size_t)mrow * Ncols + ncol] = val;
                }
            }
        }
    }
}

// ---------------------------------------------------------------------------
// Flash-style causal MQA attention.
// Grid: B * (N/16) blocks of 512 threads = 16 waves; wave w handles head w.
// All 16 heads share one (b, 16-query) block -> K/V chunks staged once in LDS.
// ---------------------------------------------------------------------------
__global__ __launch_bounds__(512)
void mqa_attn(const unsigned short* __restrict__ qb,
              const unsigned short* __restrict__ kb,
              const unsigned short* __restrict__ vb,
              unsigned short* __restrict__ ob)
{
    __shared__ __align__(16) unsigned short Kl[32 * 136];       // [key][d]
    __shared__ __align__(16) unsigned short Vl[128 * 40];       // [d][key]
    __shared__ __align__(16) unsigned short Pl[16 * 16 * 40];   // per-wave P tile

    const int tid  = threadIdx.x;
    const int lane = tid & 31;
    const int h    = tid >> 5;
    const int half = lane >> 4;
    const int l16  = lane & 15;

    const int b  = blockIdx.x >> 7;
    const int q0 = (blockIdx.x & 127) << 4;

    v16bf qf[4];
    #pragma unroll
    for (int kk = 0; kk < 4; ++kk) {
        const unsigned short* p =
            qb + (size_t)(b * 2048 + q0 + l16) * 2048 + h * 128 + kk * 32 + half * 8;
        qf[kk] = load_frag(p, p + 16);
    }

    v8f acc[8];
    #pragma unroll
    for (int i = 0; i < 8; ++i)
        #pragma unroll
        for (int e = 0; e < 8; ++e) acc[i][e] = 0.0f;
    float mi[8], li[8];
    #pragma unroll
    for (int r = 0; r < 8; ++r) { mi[r] = -1e30f; li[r] = 0.0f; }

    const float scale = 0.088388347648318447f;   // 1/sqrt(128)
    unsigned short* Pw = &Pl[h * 16 * 40];

    for (int kc = 0; kc <= q0 + 15; kc += 32) {
        __syncthreads();
        {   // cooperative K/V staging (32 keys x 128 d each)
            int kloc = tid >> 4;
            int d0   = (tid & 15) << 3;
            int key  = kc + kloc;
            U4 kraw, vraw;
            if (key < 2048) {
                kraw = *(const U4*)(kb + (size_t)(b * 2048 + key) * 128 + d0);
                vraw = *(const U4*)(vb + (size_t)(b * 2048 + key) * 128 + d0);
            } else {
                kraw.x = kraw.y = kraw.z = kraw.w = 0u;
                vraw = kraw;
            }
            *(U4*)&Kl[kloc * 136 + d0] = kraw;
            union { U4 u; unsigned short s[8]; } vv; vv.u = vraw;
            #pragma unroll
            for (int j = 0; j < 8; ++j) Vl[(d0 + j) * 40 + kloc] = vv.s[j];
        }
        __syncthreads();

        // S = Q K^T (16 x 32)
        v8f sa[2];
        #pragma unroll
        for (int nt = 0; nt < 2; ++nt)
            #pragma unroll
            for (int e = 0; e < 8; ++e) sa[nt][e] = 0.0f;
        #pragma unroll
        for (int kk = 0; kk < 4; ++kk) {
            #pragma unroll
            for (int nt = 0; nt < 2; ++nt) {
                const unsigned short* p = &Kl[(nt * 16 + l16) * 136 + kk * 32 + half * 8];
                v16bf kf = load_frag(p, p + 16);
                sa[nt] = wmma_bf16(qf[kk], kf, sa[nt]);
            }
        }

        // causal mask + online softmax (rows live in 16-lane halves)
        float alpha[8];
        #pragma unroll
        for (int r = 0; r < 8; ++r) {
            int q  = q0 + r + half * 8;
            int k0 = kc + l16;
            int k1 = kc + 16 + l16;
            float a = sa[0][r] * scale; if (k0 > q) a = -1e30f;
            float c = sa[1][r] * scale; if (k1 > q) c = -1e30f;
            float mm = fmaxf(a, c);
            mm = fmaxf(mm, __shfl_xor(mm, 1));
            mm = fmaxf(mm, __shfl_xor(mm, 2));
            mm = fmaxf(mm, __shfl_xor(mm, 4));
            mm = fmaxf(mm, __shfl_xor(mm, 8));
            float mn = fmaxf(mi[r], mm);
            alpha[r] = __expf(mi[r] - mn);
            mi[r] = mn;
            float p0 = __expf(a - mn);
            float p1 = __expf(c - mn);
            Pw[(r + half * 8) * 40 + l16]      = f2bf(p0);
            Pw[(r + half * 8) * 40 + 16 + l16] = f2bf(p1);
            float ps = p0 + p1;
            ps += __shfl_xor(ps, 1);
            ps += __shfl_xor(ps, 2);
            ps += __shfl_xor(ps, 4);
            ps += __shfl_xor(ps, 8);
            li[r] = li[r] * alpha[r] + ps;
        }
        #pragma unroll
        for (int ds = 0; ds < 8; ++ds)
            #pragma unroll
            for (int r = 0; r < 8; ++r) acc[ds][r] *= alpha[r];

        asm volatile("s_wait_dscnt 0x0" ::: "memory");  // same-wave DS in-order

        // O += P V (16 x 128)
        const unsigned short* pp = &Pw[l16 * 40 + half * 8];
        v16bf pf = load_frag(pp, pp + 16);
        #pragma unroll
        for (int ds = 0; ds < 8; ++ds) {
            const unsigned short* vp = &Vl[(ds * 16 + l16) * 40 + half * 8];
            v16bf vf = load_frag(vp, vp + 16);
            acc[ds] = wmma_bf16(pf, vf, acc[ds]);
        }
    }

    #pragma unroll
    for (int r = 0; r < 8; ++r) {
        int q = q0 + r + half * 8;
        float inv = 1.0f / li[r];
        size_t base = (size_t)(b * 2048 + q) * 2048 + h * 128;
        #pragma unroll
        for (int ds = 0; ds < 8; ++ds)
            ob[base + ds * 16 + l16] = f2bf(acc[ds][r] * inv);
    }
}

// ---------------------------------------------------------------------------
extern "C" void kernel_launch(void* const* d_in, const int* in_sizes, int n_in,
                              void* d_out, int out_size, void* d_ws, size_t ws_size,
                              hipStream_t stream)
{
    (void)in_sizes; (void)n_in; (void)out_size; (void)ws_size;

    const float* x     = (const float*)d_in[0];
    const float* w_qkv = (const float*)d_in[1];
    const float* b_qkv = (const float*)d_in[2];
    const float* w_fc  = (const float*)d_in[3];
    const float* b_fc  = (const float*)d_in[4];

    unsigned short* q_bf = (unsigned short*)d_ws;                 // 8192 x 2048
    unsigned short* k_bf = q_bf + (size_t)8192 * 2048;            // 8192 x 128
    unsigned short* v_bf = k_bf + (size_t)8192 * 128;             // 8192 x 128
    unsigned short* o_bf = v_bf + (size_t)8192 * 128;             // 8192 x 2048
    unsigned short* x_bf = o_bf + (size_t)8192 * 2048;            // 8192 x 2048
    unsigned short* wqT  = x_bf + (size_t)8192 * 2048;            // 2304 x 2048
    unsigned short* wfT  = wqT  + (size_t)2304 * 2048;            // 2048 x 2048

    // prepass: bf16 conversion + weight transposes (TDM wants [N][K] row-major)
    cvt_bf16<<<dim3((8192 * 2048) / (256 * 4)), 256, 0, stream>>>(x, x_bf);
    transpose_bf16<<<dim3(72, 64), 256, 0, stream>>>(w_qkv, wqT, 2048, 2304);
    transpose_bf16<<<dim3(64, 64), 256, 0, stream>>>(w_fc,  wfT, 2048, 2048);

    // 1) QKV GEMM + bias, split outputs to bf16 q/k/v  (TDM-fed)
    gemm_wmma_tdm<true><<<dim3(18, 64), 256, 0, stream>>>(
        x_bf, wqT, b_qkv, 2048, 2304, nullptr, q_bf, k_bf, v_bf);

    // 2) causal flash MQA attention (16 heads per WG share K/V staging)
    mqa_attn<<<dim3(512), 512, 0, stream>>>(q_bf, k_bf, v_bf, o_bf);

    // 3) FC GEMM + bias -> f32 output  (TDM-fed)
    gemm_wmma_tdm<false><<<dim3(16, 64), 256, 0, stream>>>(
        o_bf, wfT, b_fc, 2048, 2048, (float*)d_out,
        nullptr, nullptr, nullptr);
}
